// ROIHead_90821378441708
// MI455X (gfx1250) — matrix operand order repaired and checked by reading, not compiled, not run on hardware
//
#include <hip/hip_runtime.h>
#include <hip/hip_bf16.h>
#include <math.h>

typedef __attribute__((ext_vector_type(16))) __bf16 v16bf;
typedef __attribute__((ext_vector_type(8)))  float  v8f;

#define NCLS 21
#define PBINS 7
#define MAXK 6
#define FH 32
#define FW 32
#define FC 256
#define KDIM (FC*PBINS*PBINS)     /* 12544 */
#define DDIM 1024
#define NROI 300
#define MPAD 320
#define NBOX (NROI*(NCLS-1))      /* 6000 */
#define SORTN 8192
#define SCORE_THR 0.05f
#define NMS_THR 0.5f
#define TOPK 100
#define MIN_SIZE 16.0f
#define BBOX_CLIPV 4.135166556742356f

#define KSTEP 64
#define LDK 72     /* halves per LDS row: 144B, 16B-aligned, 36 banks */

// ---------------------------------------------------------------------------
// 1) ROI max-pool -> bf16 activations  X[roi][c*49 + py*7 + px]
// ---------------------------------------------------------------------------
__global__ __launch_bounds__(256) void roi_pool_k(const float* __restrict__ feat,
                                                  const float* __restrict__ prop,
                                                  const int* __restrict__ ihp,
                                                  __bf16* __restrict__ X)
{
    int roi = blockIdx.x;
    float ih = (float)ihp[0];
    float scale = exp2f(rintf(log2f((float)FH / ih)));
    float4 pr = ((const float4*)prop)[roi];
    int x1 = (int)rintf(pr.x * scale);
    int y1 = (int)rintf(pr.y * scale);
    int x2 = (int)rintf(pr.z * scale);
    int y2 = (int)rintf(pr.w * scale);
    float rw = (float)max(x2 - x1 + 1, 1);
    float rh = (float)max(y2 - y1 + 1, 1);
    for (int o = threadIdx.x; o < KDIM; o += 256) {
        int c = o / (PBINS*PBINS);
        int bin = o % (PBINS*PBINS);
        int py = bin / PBINS, px = bin % PBINS;
        int hs  = min(max((int)floorf(py * rh / (float)PBINS) + y1, 0), FH);
        int he  = min(max((int)ceilf((py + 1.0f) * rh / (float)PBINS) + y1, 0), FH);
        int wsb = min(max((int)floorf(px * rw / (float)PBINS) + x1, 0), FW);
        int we  = min(max((int)ceilf((px + 1.0f) * rw / (float)PBINS) + x1, 0), FW);
        float v = 0.0f;
        if (he > hs && we > wsb) {
            float m = -INFINITY;
            for (int i = 0; i < MAXK; ++i) {          // KB=6 truncation as in reference
                int hh = hs + i; if (hh >= he) break;
                const float* row = feat + c*FH*FW + hh*FW;
                for (int j = 0; j < MAXK; ++j) {
                    int ww = wsb + j; if (ww >= we) break;
                    m = fmaxf(m, row[ww]);
                }
            }
            v = m;
        }
        X[(size_t)roi*KDIM + o] = (__bf16)v;
    }
}

// ---------------------------------------------------------------------------
// 2) GEMM: bf16 A [MPADxK] @ f32 W [KxN] (+bias, relu?) -> f32 / bf16
//    256 thr = 8 waves; macro-tile 32(M) x 128(N); K-step 64; double-buffered
//    LDS; 4 v_wmma_f32_16x16x32_bf16 per wave per iteration; 1 barrier/iter.
//    B loads are clamped+selected (no exec-mask predication in hot loop).
// ---------------------------------------------------------------------------
__global__ __launch_bounds__(256) void gemm_bf16_wmma_k(
    const __bf16* __restrict__ A, int K,
    const float*  __restrict__ W, int N,
    const float*  __restrict__ bias,
    float* __restrict__ outF, __bf16* __restrict__ outB,
    int relu, int Mstore)
{
    __shared__ __bf16 Ash[2][32][LDK];    // M x K halves
    __shared__ __bf16 Bsh[2][128][LDK];   // N x K halves (transposed)
    const int tid  = threadIdx.x;
    const int lane = tid & 31;
    const int wv   = tid >> 5;
    const int wm   = wv >> 2;             // 0..1  -> M subtile
    const int wn   = wv & 3;              // 0..3  -> N subtile pair (32 wide)
    const int m0 = blockIdx.y * 32;
    const int n0 = blockIdx.x * 128;

    // staging mapping
    const int ar  = tid >> 3;             // A row 0..31
    const int ac  = (tid & 7) * 8;        // A col (halves), mult of 8
    const int bn  = tid & 127;            // B col 0..127
    const int bkg = (tid >> 7) * 32;      // B k-group 0 or 32
    const bool bok = (n0 + bn) < N;
    const int  bncl = min(n0 + bn, N - 1);          // clamped: always in-range
    const float* __restrict__ Wp = W + bncl;        // column base pointer
    const __bf16* __restrict__ Ap = A + (size_t)(m0 + ar)*K + ac;
    const int kTiles = K / KSTEP;

    uint4 areg;
    float breg[32];

    // ---- prologue: load + store tile 0 ----
    areg = *(const uint4*)Ap;
    {
        const float* p = Wp + (size_t)bkg * N;
        #pragma unroll
        for (int j = 0; j < 32; ++j) { float v = *p; breg[j] = bok ? v : 0.0f; p += N; }
    }
    *(uint4*)&Ash[0][ar][ac] = areg;
    #pragma unroll
    for (int g = 0; g < 8; ++g) {
        union { uint2 u2; __bf16 h[4]; } pk;
        pk.h[0] = (__bf16)breg[g*4+0]; pk.h[1] = (__bf16)breg[g*4+1];
        pk.h[2] = (__bf16)breg[g*4+2]; pk.h[3] = (__bf16)breg[g*4+3];
        *(uint2*)&Bsh[0][bn][bkg + g*4] = pk.u2;
    }
    __syncthreads();

    v8f acc0 = {}, acc1 = {};
    const int ah = (lane >> 4) << 3;      // A uint offset in 16-uint half: 0 / 8
    const int bh = (lane >> 4) << 3;      // B uint offset in 16-uint half: 0 / 8

    for (int kt = 0; kt < kTiles; ++kt) {
        const int cur = kt & 1, nxt = cur ^ 1;
        const bool more = (kt + 1) < kTiles;
        const int kk = (kt + 1) * KSTEP;
        // issue next tile's global loads first (overlap with WMMAs below)
        if (more) {
            areg = *(const uint4*)(Ap + kk);
            const float* p = Wp + (size_t)(kk + bkg) * N;
            #pragma unroll
            for (int j = 0; j < 32; ++j) { float v = *p; breg[j] = bok ? v : 0.0f; p += N; }
        }
        // compute from current buffer: 2 K-halves x 2 N-subtiles = 4 WMMA
        const unsigned int* arp  = (const unsigned int*)&Ash[cur][wm*16 + (lane & 15)][0];
        const unsigned int* brp0 = (const unsigned int*)&Bsh[cur][wn*32 +      (lane & 15)][0];
        const unsigned int* brp1 = (const unsigned int*)&Bsh[cur][wn*32 + 16 + (lane & 15)][0];
        #pragma unroll
        for (int s = 0; s < 2; ++s) {
            union { v16bf v; unsigned int u[8]; } af, bf0, bf1;
            const int ab = s*16 + ah;
            #pragma unroll
            for (int p = 0; p < 4; ++p) { af.u[p] = arp[ab + p]; af.u[4+p] = arp[ab + 4 + p]; }
            const int bb = s*16 + bh;
            #pragma unroll
            for (int p = 0; p < 8; ++p) { bf0.u[p] = brp0[bb + p]; bf1.u[p] = brp1[bb + p]; }
            acc0 = __builtin_amdgcn_wmma_f32_16x16x32_bf16(false, af.v, false, bf0.v,
                                                           (short)0, acc0, false, false);
            acc1 = __builtin_amdgcn_wmma_f32_16x16x32_bf16(false, af.v, false, bf1.v,
                                                           (short)0, acc1, false, false);
        }
        // store next tile into the other buffer
        if (more) {
            *(uint4*)&Ash[nxt][ar][ac] = areg;
            #pragma unroll
            for (int g = 0; g < 8; ++g) {
                union { uint2 u2; __bf16 h[4]; } pk;
                pk.h[0] = (__bf16)breg[g*4+0]; pk.h[1] = (__bf16)breg[g*4+1];
                pk.h[2] = (__bf16)breg[g*4+2]; pk.h[3] = (__bf16)breg[g*4+3];
                *(uint2*)&Bsh[nxt][bn][bkg + g*4] = pk.u2;
            }
        }
        __syncthreads();
    }

    // epilogue per C/D layout: vgpr r -> m = r + 8*(lane>=16); n = lane&15
    const int mb = m0 + wm*16 + ((lane >> 4) << 3);
    #pragma unroll
    for (int c = 0; c < 2; ++c) {
        const int n = n0 + wn*32 + c*16 + (lane & 15);
        const float bv = (bias && n < N) ? bias[n] : 0.0f;
        const v8f acc = c ? acc1 : acc0;
        #pragma unroll
        for (int r = 0; r < 8; ++r) {
            const int m = mb + r;
            float v = acc[r] + bv;
            if (relu) v = fmaxf(v, 0.0f);
            if (n < N && m < Mstore) {
                if (outF) outF[(size_t)m*N + n] = v;
                if (outB) outB[(size_t)m*N + n] = (__bf16)v;
            }
        }
    }
}

// ---------------------------------------------------------------------------
// 3) Post-process: softmax, decode, sort, greedy NMS, top-100  (1 block)
// ---------------------------------------------------------------------------
__global__ __launch_bounds__(1024) void post_k(
    const float* __restrict__ prop,
    const float* __restrict__ CLS,    // [300][21] logits
    const float* __restrict__ REG,    // [300][84]
    const int* __restrict__ ihp, const int* __restrict__ iwp,
    float* __restrict__ SPROB,        // [300][21]
    float* __restrict__ BX,           // [6000][4] clipped boxes
    float* __restrict__ SC,           // [6000]
    float* __restrict__ out)          // 600: boxes[400] labels[100](i32 bits) scores[100]
{
    __shared__ float s_key[SORTN];
    __shared__ int   s_idx[SORTN];
    __shared__ unsigned char s_keep[SORTN];
    int tid = threadIdx.x;
    float ih = (float)ihp[0], iw = (float)iwp[0];

    for (int t = tid; t < 600; t += 1024) out[t] = 0.0f;

    for (int r = tid; r < NROI; r += 1024) {               // softmax rows
        float mx = -INFINITY;
        for (int c = 0; c < NCLS; ++c) mx = fmaxf(mx, CLS[r*NCLS + c]);
        float s = 0.0f;
        for (int c = 0; c < NCLS; ++c) s += expf(CLS[r*NCLS + c] - mx);
        float inv = 1.0f / s;
        for (int c = 0; c < NCLS; ++c) SPROB[r*NCLS + c] = expf(CLS[r*NCLS + c] - mx) * inv;
    }
    __syncthreads();

    for (int g = tid; g < NBOX; g += 1024) {               // decode + clip + valid
        int roi = g / (NCLS - 1);
        int c = g % (NCLS - 1) + 1;
        float px1 = prop[roi*4+0], py1 = prop[roi*4+1];
        float px2 = prop[roi*4+2], py2 = prop[roi*4+3];
        float w = px2 - px1, h = py2 - py1;
        float cx = px1 + 0.5f*w, cy = py1 + 0.5f*h;
        float dx = REG[roi*84 + c*4+0], dy = REG[roi*84 + c*4+1];
        float dw = fminf(REG[roi*84 + c*4+2], BBOX_CLIPV);
        float dh = fminf(REG[roi*84 + c*4+3], BBOX_CLIPV);
        float pcx = dx*w + cx, pcy = dy*h + cy;
        float pw = expf(dw)*w, ph = expf(dh)*h;
        float x1 = fminf(fmaxf(pcx - 0.5f*pw, 0.0f), iw);
        float y1 = fminf(fmaxf(pcy - 0.5f*ph, 0.0f), ih);
        float x2 = fminf(fmaxf(pcx + 0.5f*pw, 0.0f), iw);
        float y2 = fminf(fmaxf(pcy + 0.5f*ph, 0.0f), ih);
        float sc = SPROB[roi*NCLS + c];
        BX[g*4+0]=x1; BX[g*4+1]=y1; BX[g*4+2]=x2; BX[g*4+3]=y2;
        SC[g] = sc;
        bool valid = (sc > SCORE_THR) && ((x2-x1) >= MIN_SIZE) && ((y2-y1) >= MIN_SIZE);
        s_key[g] = valid ? sc : -INFINITY;
        s_idx[g] = g;
    }
    for (int g = NBOX + tid; g < SORTN; g += 1024) { s_key[g] = -INFINITY; s_idx[g] = g; }

    // bitonic sort: score desc, index asc (stable-argsort emulation)
    for (int size = 2; size <= SORTN; size <<= 1) {
        for (int stride = size >> 1; stride > 0; stride >>= 1) {
            __syncthreads();
            for (int t = tid; t < SORTN/2; t += 1024) {
                int pos = 2*t - (t & (stride - 1));
                int a = pos, b = pos + stride;
                float ka = s_key[a], kb2 = s_key[b];
                int   ia = s_idx[a], ib  = s_idx[b];
                bool ddd = ((pos & size) == 0);
                bool b_first = (kb2 > ka) || (kb2 == ka && ib < ia);
                bool a_first = (ka > kb2) || (ka == kb2 && ia < ib);
                bool sw = ddd ? b_first : a_first;
                if (sw) { s_key[a]=kb2; s_idx[a]=ib; s_key[b]=ka; s_idx[b]=ia; }
            }
        }
    }
    __syncthreads();
    for (int g = tid; g < SORTN; g += 1024) s_keep[g] = (g < NBOX) && (s_key[g] > 0.0f);
    float offsc = fmaxf(ih, iw) + 1.0f;

    // sequential greedy NMS (matches reference scan semantics)
    for (int i = 0; i < NBOX; ++i) {
        __syncthreads();
        if (!s_keep[i]) continue;                 // uniform: LDS read after barrier
        int gi = s_idx[i];
        float offi = (float)(gi % (NCLS-1) + 1) * offsc;
        float ax1 = BX[gi*4+0]+offi, ay1 = BX[gi*4+1]+offi;
        float ax2 = BX[gi*4+2]+offi, ay2 = BX[gi*4+3]+offi;
        float aarea = (ax2-ax1)*(ay2-ay1);
        for (int j = i + 1 + tid; j < NBOX; j += 1024) {
            if (!s_keep[j]) continue;
            int gj = s_idx[j];
            float offj = (float)(gj % (NCLS-1) + 1) * offsc;
            float bx1 = BX[gj*4+0]+offj, by1 = BX[gj*4+1]+offj;
            float bx2 = BX[gj*4+2]+offj, by2 = BX[gj*4+3]+offj;
            float ww = fmaxf(fminf(ax2,bx2) - fmaxf(ax1,bx1), 0.0f);
            float hh = fmaxf(fminf(ay2,by2) - fmaxf(ay1,by1), 0.0f);
            float inter = ww*hh;
            float uni = aarea + (bx2-bx1)*(by2-by1) - inter;
            float iou = (uni > 0.0f) ? inter/uni : 0.0f;
            if (iou > NMS_THR) s_keep[j] = 0;
        }
    }
    __syncthreads();

    if (tid == 0) {                               // emit top-100 kept (already sorted)
        int k = 0;
        for (int i = 0; i < NBOX && k < TOPK; ++i) {
            if (!s_keep[i]) continue;
            int g = s_idx[i];
            out[k*4+0]=BX[g*4+0]; out[k*4+1]=BX[g*4+1];
            out[k*4+2]=BX[g*4+2]; out[k*4+3]=BX[g*4+3];
            ((int*)(out + 4*TOPK))[k] = g % (NCLS-1) + 1;
            out[5*TOPK + k] = SC[g];
            ++k;
        }
    }
}

// ---------------------------------------------------------------------------
extern "C" void kernel_launch(void* const* d_in, const int* in_sizes, int n_in,
                              void* d_out, int out_size, void* d_ws, size_t ws_size,
                              hipStream_t stream) {
    (void)in_sizes; (void)n_in; (void)out_size; (void)ws_size;
    const float* feat = (const float*)d_in[0];
    const float* prop = (const float*)d_in[1];
    const float* W6 = (const float*)d_in[2];
    const float* b6 = (const float*)d_in[3];
    const float* W7 = (const float*)d_in[4];
    const float* b7 = (const float*)d_in[5];
    const float* Wc = (const float*)d_in[6];
    const float* bc = (const float*)d_in[7];
    const float* Wr = (const float*)d_in[8];
    const float* br = (const float*)d_in[9];
    const int* ihp = (const int*)d_in[10];
    const int* iwp = (const int*)d_in[11];

    char* wsp = (char*)d_ws;
    size_t off = 0;
    auto take = [&](size_t bytes) { char* p = wsp + off; off = (off + bytes + 255) & ~(size_t)255; return p; };
    __bf16* X6 = (__bf16*)take((size_t)MPAD * KDIM * 2);
    __bf16* X7 = (__bf16*)take((size_t)MPAD * DDIM * 2);
    __bf16* X8 = (__bf16*)take((size_t)MPAD * DDIM * 2);
    size_t zbytes = off;                       // zero activation buffers (pad rows)
    float* CLSb = (float*)take((size_t)NROI * NCLS * 4);
    float* REGb = (float*)take((size_t)NROI * (NCLS*4) * 4);
    float* SPROB = (float*)take((size_t)NROI * NCLS * 4);
    float* BX = (float*)take((size_t)NBOX * 4 * 4);
    float* SC = (float*)take((size_t)NBOX * 4);

    hipMemsetAsync(d_ws, 0, zbytes, stream);
    roi_pool_k<<<NROI, 256, 0, stream>>>(feat, prop, ihp, X6);
    gemm_bf16_wmma_k<<<dim3(DDIM/128, MPAD/32), 256, 0, stream>>>(X6, KDIM, W6, DDIM, b6, nullptr, X7, 1, NROI);
    gemm_bf16_wmma_k<<<dim3(DDIM/128, MPAD/32), 256, 0, stream>>>(X7, DDIM, W7, DDIM, b7, nullptr, X8, 1, NROI);
    gemm_bf16_wmma_k<<<dim3(1, MPAD/32), 256, 0, stream>>>(X8, DDIM, Wc, NCLS, bc, CLSb, nullptr, 0, NROI);
    gemm_bf16_wmma_k<<<dim3(1, MPAD/32), 256, 0, stream>>>(X8, DDIM, Wr, NCLS*4, br, REGb, nullptr, 0, NROI);
    post_k<<<1, 1024, 0, stream>>>(prop, CLSb, REGb, ihp, iwp, SPROB, BX, SC, (float*)d_out);
}
